// MultiHeadMixer_48137993453901
// MI455X (gfx1250) — compile-verified, matrix operand order checked by reading, not run on hardware
//
#include <hip/hip_runtime.h>
#include <hip/hip_bf16.h>
#include <math.h>

// MultiHeadMixer on gfx1250 (MI455X, wave32 WMMA):
//  - one-time f32->f16 conversion of x / inp_w / out_w (kills in-loop cvts)
//  - WMMA input projection (16x32 tile per wave)
//  - causal-Toeplitz mixing: per head a 16xSxS WMMA GEMM; the Toeplitz B
//    fragments are gathered from a REVERSED, ZERO-PADDED weight row in LDS,
//    so the causal mask costs zero VALU (pad reads return 0)
//  - WMMA output projection (16x32 tile per wave), f32 coalesced stores

#define BB 2
#define SS 2048
#define EE 512
#define HH 32
#define DD 16

typedef __attribute__((ext_vector_type(16))) _Float16 v16h;
typedef __attribute__((ext_vector_type(8)))  _Float16 v8h;
typedef __attribute__((ext_vector_type(8)))  float    v8f;

__device__ __forceinline__ float softplusf(float x) {
    return x > 20.f ? x : log1pf(expf(x));
}

// ---------------------------------------------------------------------------
// f32 -> f16 elementwise (8 elems/thread, 32B in / 16B out per thread)
// ---------------------------------------------------------------------------
__global__ __launch_bounds__(256) void k_cvt(const float* __restrict__ in,
                                             _Float16* __restrict__ out, int n8) {
    int i = blockIdx.x * 256 + threadIdx.x;
    if (i >= n8) return;
    v8f v = *(const v8f*)(in + (size_t)i * 8);
    v8h o;
#pragma unroll
    for (int e = 0; e < 8; ++e) o[e] = (_Float16)v[e];
    *(v8h*)(out + (size_t)i * 8) = o;
}

// ---------------------------------------------------------------------------
// w = softplus(weight_raw) (f16), ninv = 1/cumsum(w) (f32).
// One block per head; 256 threads x 8 elements; Hillis-Steele scan of totals.
// ---------------------------------------------------------------------------
__global__ __launch_bounds__(256) void k_prep(const float* __restrict__ raw,
                                              _Float16* __restrict__ w_h,
                                              float* __restrict__ ninv) {
    __shared__ float tsum[256];
    const int h = blockIdx.x, tid = threadIdx.x;
    const float* r = raw + (size_t)h * SS + tid * 8;
    float pre[8];
    float run = 0.f;
#pragma unroll
    for (int i = 0; i < 8; ++i) {
        float spv = softplusf(r[i]);
        w_h[(size_t)h * SS + tid * 8 + i] = (_Float16)spv;
        run += spv;
        pre[i] = run;
    }
    tsum[tid] = run;
    __syncthreads();
    for (int off = 1; off < 256; off <<= 1) {
        float t = (tid >= off) ? tsum[tid - off] : 0.f;
        __syncthreads();
        tsum[tid] += t;
        __syncthreads();
    }
    const float excl = tsum[tid] - run;
#pragma unroll
    for (int i = 0; i < 8; ++i)
        ninv[(size_t)h * SS + tid * 8 + i] = 1.f / (excl + pre[i]);
}

// ---------------------------------------------------------------------------
// Input projection: C[row,n] = sum_k x[row,k]*inp_w[n,k] + inp_b[n].
// All-f16 operands; each wave computes a 16x32 tile (2 WMMAs / K-chunk).
// Output hproj in (B, E, S) layout so the mixer's K runs are contiguous.
// ---------------------------------------------------------------------------
__global__ __launch_bounds__(128) void k_proj_in(const _Float16* __restrict__ xh,
                                                 const _Float16* __restrict__ Wh,
                                                 const float* __restrict__ bvec,
                                                 _Float16* __restrict__ hproj) {
    const int lane = threadIdx.x & 31;
    const int wv   = threadIdx.x >> 5;
    const int r0   = blockIdx.x * 16;                 // M tile (rows = b*S+s)
    const int n0   = (blockIdx.y * 4 + wv) * 32;      // N tile (output feature)
    const int nn   = lane & 15;
    const int kb   = (lane < 16) ? 0 : 8;             // per-lane K base (ISA layout)
    const _Float16* arow  = xh + (size_t)(r0 + nn) * EE;
    const _Float16* brow0 = Wh + (size_t)(n0 + nn) * EE;
    const _Float16* brow1 = Wh + (size_t)(n0 + 16 + nn) * EE;
    v8f acc0 = {}, acc1 = {};
    for (int k0 = 0; k0 < EE; k0 += 32) {
        v8h a0 = *(const v8h*)(arow + k0 + kb);
        v8h a1 = *(const v8h*)(arow + k0 + 16 + kb);
        v8h p0 = *(const v8h*)(brow0 + k0 + kb);
        v8h p1 = *(const v8h*)(brow0 + k0 + 16 + kb);
        v8h q0 = *(const v8h*)(brow1 + k0 + kb);
        v8h q1 = *(const v8h*)(brow1 + k0 + 16 + kb);
        v16h a, b0, b1;
#pragma unroll
        for (int i = 0; i < 8; ++i) {
            a[i] = a0[i];  a[i + 8] = a1[i];
            b0[i] = p0[i]; b0[i + 8] = p1[i];
            b1[i] = q0[i]; b1[i + 8] = q1[i];
        }
        acc0 = __builtin_amdgcn_wmma_f32_16x16x32_f16(false, a, false, b0,
                                                      (short)0, acc0, false, false);
        acc1 = __builtin_amdgcn_wmma_f32_16x16x32_f16(false, a, false, b1,
                                                      (short)0, acc1, false, false);
    }
    const int   mofs = (lane < 16) ? 0 : 8;
    const int   na   = n0 + nn;
    const int   nb   = n0 + 16 + nn;
    const float ba   = bvec[na];
    const float bb   = bvec[nb];
#pragma unroll
    for (int v = 0; v < 8; ++v) {
        int row = r0 + v + mofs;
        int b_  = row >> 11;          // / S
        int s   = row & (SS - 1);     // % S
        hproj[((size_t)(b_ * EE + na)) * SS + s] = (_Float16)(acc0[v] + ba);
        hproj[((size_t)(b_ * EE + nb)) * SS + s] = (_Float16)(acc1[v] + bb);
    }
}

// ---------------------------------------------------------------------------
// Causal Toeplitz mixing. Per head h: Y(16 x S) = Hh(16 x S) * M,
// M[s,t] = w[h, t-s] for t>=s. The head's weight row is staged REVERSED and
// ZERO-PADDED in LDS: wlr[i] = w[S-1-i], wlr[S..S+63] = 0. Fragment index
// i = (S-1-t) + s0 + kk is increasing in kk and lands in the zero pad exactly
// when causality demands a zero -> unmasked ds_load_u16 gathers, no selects.
// Both batches share each synthesized B fragment (2 WMMAs per fragment).
// ---------------------------------------------------------------------------
__global__ __launch_bounds__(128) void k_mix(const _Float16* __restrict__ w_h,
                                             const float* __restrict__ ninv,
                                             const float* __restrict__ bias,
                                             const _Float16* __restrict__ hproj,
                                             _Float16* __restrict__ ymid) {
    __shared__ __align__(16) _Float16 wlr[SS + 64];
    const int lane = threadIdx.x & 31;
    const int wv   = threadIdx.x >> 5;
    const int h    = blockIdx.y;
    const int t0   = blockIdx.x * 64 + wv * 16;
    // cooperative reversed preload + zero pad
    if (threadIdx.x < 64) wlr[SS + threadIdx.x] = (_Float16)0.f;
    for (int j = threadIdx.x; j < SS / 8; j += 128) {
        v8h g = *(const v8h*)(w_h + (size_t)h * SS + j * 8);
        v8h rg;
#pragma unroll
        for (int e = 0; e < 8; ++e) rg[e] = g[7 - e];
        *(v8h*)(wlr + (SS - 8 - j * 8)) = rg;
    }
    __syncthreads();

    const int nn = lane & 15;
    const int kb = (lane < 16) ? 0 : 8;
    const int t  = t0 + nn;
    const _Float16* arow0 = hproj + ((size_t)(0 * EE) + h * DD + nn) * SS;
    const _Float16* arow1 = hproj + ((size_t)(1 * EE) + h * DD + nn) * SS;
    v8f acc0 = {}, acc1 = {};
    for (int s0 = 0; s0 < t0 + 16; s0 += 32) {   // causal K taper
        __builtin_prefetch(arow0 + s0 + 128, 0, 1);   // global_prefetch_b8
        __builtin_prefetch(arow1 + s0 + 128, 0, 1);
        v8h x0 = *(const v8h*)(arow0 + s0 + kb);
        v8h x1 = *(const v8h*)(arow0 + s0 + 16 + kb);
        v8h y0 = *(const v8h*)(arow1 + s0 + kb);
        v8h y1 = *(const v8h*)(arow1 + s0 + 16 + kb);
        v16h a0, a1, bm;
#pragma unroll
        for (int i = 0; i < 8; ++i) {
            a0[i] = x0[i]; a0[i + 8] = x1[i];
            a1[i] = y0[i]; a1[i + 8] = y1[i];
        }
        const int ib = (SS - 1 - t) + s0 + kb;
#pragma unroll
        for (int r = 0; r < 8; ++r) {
            bm[r]     = wlr[ib + r];         // K = kb + r
            bm[8 + r] = wlr[ib + 16 + r];    // K = 16 + kb + r
        }
        acc0 = __builtin_amdgcn_wmma_f32_16x16x32_f16(false, a0, false, bm,
                                                      (short)0, acc0, false, false);
        acc1 = __builtin_amdgcn_wmma_f32_16x16x32_f16(false, a1, false, bm,
                                                      (short)0, acc1, false, false);
    }
    const float sc  = ninv[(size_t)h * SS + t];
    const float bv  = bias[(size_t)h * SS + t];
    const int  mofs = (lane < 16) ? 0 : 8;
    v8h o0, o1;
#pragma unroll
    for (int v = 0; v < 8; ++v) {
        o0[v] = (_Float16)(acc0[v] * sc + bv);
        o1[v] = (_Float16)(acc1[v] * sc + bv);
    }
    // ymid layout (B, S, E): contiguous 16B store of 8 head-dims per lane
    *(v8h*)(ymid + ((size_t)(0 * SS + t)) * EE + h * DD + mofs) = o0;
    *(v8h*)(ymid + ((size_t)(1 * SS + t)) * EE + h * DD + mofs) = o1;
}

// ---------------------------------------------------------------------------
// Output projection: out[row,n] = sum_k ymid[row,k]*out_w[n,k].
// All-f16 operands; 16x32 tile per wave; f32 coalesced stores.
// ---------------------------------------------------------------------------
__global__ __launch_bounds__(128) void k_proj_out(const _Float16* __restrict__ ymid,
                                                  const _Float16* __restrict__ Wh,
                                                  float* __restrict__ out) {
    const int lane = threadIdx.x & 31;
    const int wv   = threadIdx.x >> 5;
    const int r0   = blockIdx.x * 16;
    const int n0   = (blockIdx.y * 4 + wv) * 32;
    const int nn   = lane & 15;
    const int kb   = (lane < 16) ? 0 : 8;
    const _Float16* arow  = ymid + (size_t)(r0 + nn) * EE;
    const _Float16* brow0 = Wh + (size_t)(n0 + nn) * EE;
    const _Float16* brow1 = Wh + (size_t)(n0 + 16 + nn) * EE;
    v8f acc0 = {}, acc1 = {};
    for (int k0 = 0; k0 < EE; k0 += 32) {
        v8h a0 = *(const v8h*)(arow + k0 + kb);
        v8h a1 = *(const v8h*)(arow + k0 + 16 + kb);
        v8h p0 = *(const v8h*)(brow0 + k0 + kb);
        v8h p1 = *(const v8h*)(brow0 + k0 + 16 + kb);
        v8h q0 = *(const v8h*)(brow1 + k0 + kb);
        v8h q1 = *(const v8h*)(brow1 + k0 + 16 + kb);
        v16h a, b0, b1;
#pragma unroll
        for (int i = 0; i < 8; ++i) {
            a[i] = a0[i];  a[i + 8] = a1[i];
            b0[i] = p0[i]; b0[i + 8] = p1[i];
            b1[i] = q0[i]; b1[i + 8] = q1[i];
        }
        acc0 = __builtin_amdgcn_wmma_f32_16x16x32_f16(false, a, false, b0,
                                                      (short)0, acc0, false, false);
        acc1 = __builtin_amdgcn_wmma_f32_16x16x32_f16(false, a, false, b1,
                                                      (short)0, acc1, false, false);
    }
    const int mofs = (lane < 16) ? 0 : 8;
#pragma unroll
    for (int v = 0; v < 8; ++v) {
        out[(size_t)(r0 + v + mofs) * EE + n0 + nn]      = acc0[v];
        out[(size_t)(r0 + v + mofs) * EE + n0 + 16 + nn] = acc1[v];
    }
}

// ---------------------------------------------------------------------------
extern "C" void kernel_launch(void* const* d_in, const int* in_sizes, int n_in,
                              void* d_out, int out_size, void* d_ws, size_t ws_size,
                              hipStream_t stream) {
    const float* x     = (const float*)d_in[0];   // (B,S,E)
    const float* wraw  = (const float*)d_in[1];   // (H,S)
    const float* bias  = (const float*)d_in[2];   // (H,S)
    const float* inp_w = (const float*)d_in[3];   // (E,E)
    const float* inp_b = (const float*)d_in[4];   // (E,)
    const float* out_w = (const float*)d_in[5];   // (E,E)
    float*       out   = (float*)d_out;           // (B,S,E)

    // workspace carve-up (all slab offsets 256B aligned)
    char* ws = (char*)d_ws;
    _Float16* w_h   = (_Float16*)ws;                      // 128 KB
    float*    ninv  = (float*)(ws + 131072);              // 256 KB
    _Float16* hproj = (_Float16*)(ws + 393216);           // 4 MB  (B,E,S) f16
    _Float16* ymid  = (_Float16*)(ws + 4587520);          // 4 MB  (B,S,E) f16
    _Float16* x_h   = (_Float16*)(ws + 8781824);          // 4 MB  x in f16
    _Float16* wih   = (_Float16*)(ws + 12976128);         // 512 KB inp_w f16
    _Float16* woh   = (_Float16*)(ws + 13500416);         // 512 KB out_w f16

    k_cvt<<<dim3((BB * SS * EE / 8 + 255) / 256), dim3(256), 0, stream>>>(x, x_h, BB * SS * EE / 8);
    k_cvt<<<dim3((EE * EE / 8 + 255) / 256), dim3(256), 0, stream>>>(inp_w, wih, EE * EE / 8);
    k_cvt<<<dim3((EE * EE / 8 + 255) / 256), dim3(256), 0, stream>>>(out_w, woh, EE * EE / 8);
    k_prep<<<dim3(HH), dim3(256), 0, stream>>>(wraw, w_h, ninv);
    k_proj_in<<<dim3((BB * SS) / 16, EE / 128), dim3(128), 0, stream>>>(x_h, wih, inp_b, hproj);
    k_mix<<<dim3(SS / 64, HH), dim3(128), 0, stream>>>(w_h, ninv, bias, hproj, ymid);
    k_proj_out<<<dim3((BB * SS) / 16, EE / 128), dim3(128), 0, stream>>>(ymid, woh, out);
}
